// MultiHeadedAttentionLayer_46377056862230
// MI455X (gfx1250) — matrix operand order repaired and checked by reading, not compiled, not run on hardware
//
#include <hip/hip_runtime.h>
#include <cstdint>

#define B_  2
#define S_  4096
#define DM_ 512
#define H_  8
#define D_  64
#define BS_ 64
#define NB_ 64
#define R_  3

typedef __attribute__((ext_vector_type(16))) __bf16 v16bf;
typedef __attribute__((ext_vector_type(8)))  float  v8f;

union frag_q2 { uint4 q[2]; v16bf v; };        // two 16B chunks -> 16 bf16
union pack_u  { uint32_t d[4]; uint4 q; };

__device__ __forceinline__ unsigned short f2bf_bits(float f) {
    unsigned u = __float_as_uint(f);
    unsigned r = (u + 0x7FFFu + ((u >> 16) & 1u)) >> 16;   // RNE
    return (unsigned short)r;
}
__device__ __forceinline__ uint32_t pack2bf(float lo, float hi) {
    return (uint32_t)f2bf_bits(lo) | ((uint32_t)f2bf_bits(hi) << 16);
}

// ---- fragment loaders (wave32 CDNA5 16-bit layouts, CDNA5 ISA) ----

// A-matrix 16x32 from bf16 row-major: lane<16 holds K 0..7 / 16..23,
// lane>=16 holds K 8..15 / 24..31.  Two aligned 16B loads per lane.
__device__ __forceinline__ v16bf load_a_bf16(const unsigned short* base, int stride,
                                             int row, int kc, int lane) {
    const unsigned short* p = base + (size_t)row * stride + kc + ((lane < 16) ? 0 : 8);
    frag_q2 f;
    f.q[0] = *(const uint4*)p;
    f.q[1] = *(const uint4*)(p + 16);
    return f.v;
}
// B-matrix 32x16 where B[k][n] = src[nrow(n)*stride + kc + k] (contiguous in k):
// lane<16 holds K 0..7 / 16..23, lane>=16 holds K 8..15 / 24..31.  Two aligned 16B loads.
__device__ __forceinline__ v16bf load_bT_bf16(const unsigned short* base, int stride,
                                              int nrow, int kc, int lane) {
    const unsigned short* p = base + (size_t)nrow * stride + kc + ((lane < 16) ? 0 : 8);
    frag_q2 f;
    f.q[0] = *(const uint4*)p;
    f.q[1] = *(const uint4*)(p + 16);
    return f.v;
}

__device__ __forceinline__ v8f wmma_bf16(v16bf a, v16bf b, v8f c) {
    return __builtin_amdgcn_wmma_f32_16x16x32_bf16(false, a, false, b, (short)0, c, false, false);
}

// -------- Pre-pass A: x (fp32) -> xbf (bf16), 8 elements per thread --------
__global__ __launch_bounds__(256) void conv_x_kernel(const float* __restrict__ x,
                                                     unsigned short* __restrict__ xbf) {
    size_t i0 = ((size_t)blockIdx.x * 256 + threadIdx.x) * 8;
    float4 a = *(const float4*)(x + i0);
    float4 b = *(const float4*)(x + i0 + 4);
    pack_u p;
    p.d[0] = pack2bf(a.x, a.y);
    p.d[1] = pack2bf(a.z, a.w);
    p.d[2] = pack2bf(b.x, b.y);
    p.d[3] = pack2bf(b.z, b.w);
    *(uint4*)(xbf + i0) = p.q;
}

// ---- Pre-pass B: w (fp32 [k][n]) -> wT (bf16 [n][k]) for q,k,v weights ----
// one dword (2 k-values) per thread; grid = 3*512*256/256 blocks
__global__ __launch_bounds__(256) void conv_w_kernel(
    const float* __restrict__ wq, const float* __restrict__ wk,
    const float* __restrict__ wv, unsigned short* __restrict__ wt) {
    size_t idx = (size_t)blockIdx.x * 256 + threadIdx.x;   // [z][n][k/2]
    int z = (int)(idx >> 17);                              // 512*256 = 131072
    size_t rem = idx & 131071;
    int n = (int)(rem >> 8);
    int k0 = ((int)(rem & 255)) * 2;
    const float* w = (z == 0) ? wq : ((z == 1) ? wk : wv);
    uint32_t d = pack2bf(w[(size_t)k0 * DM_ + n], w[(size_t)(k0 + 1) * DM_ + n]);
    *(uint32_t*)(wt + (size_t)z * DM_ * DM_ + (size_t)n * DM_ + k0) = d;
}

// ------------------- Stage 1: QKV projection (bf16 WMMA) -------------------
// grid (128, 8, 3), block 128.  z: 0=Q 1=K 2=V.  y: head (64-col tile).
// Q,K layout [b][h][s][d] bf16; V stored transposed [b][h][d][s] bf16.
__global__ __launch_bounds__(128) void qkv_proj_kernel(
    const unsigned short* __restrict__ xbf,
    const unsigned short* __restrict__ wt,
    const float* __restrict__ bq, const float* __restrict__ bk,
    const float* __restrict__ bv,
    unsigned short* __restrict__ q, unsigned short* __restrict__ k,
    unsigned short* __restrict__ vt) {
    int z = blockIdx.z;
    const unsigned short* w = wt + (size_t)z * DM_ * DM_;   // [n][k]
    const float* bias = (z == 0) ? bq : ((z == 1) ? bk : bv);

    int wave = threadIdx.x >> 5, lane = threadIdx.x & 31;
    int h = blockIdx.y;
    int row0 = blockIdx.x * 64;                 // 64 rows of x (b*S + s)
    int rsel = (lane < 16) ? 0 : 8;
    int b = row0 >> 12;

    v8f acc[4] = {v8f{}, v8f{}, v8f{}, v8f{}};

    if (z < 2) {
        // ---- Q/K: A = wT (M = out dim), B = x^T view (N = seq) ----
        unsigned short* out = (z == 0) ? q : k;
        int arow = h * 64 + wave * 16 + (lane & 15);        // output column
        for (int kc = 0; kc < DM_; kc += 32) {
            v16bf af = load_a_bf16(w, DM_, arow, kc, lane);
#pragma unroll
            for (int nt = 0; nt < 4; ++nt) {
                v16bf bf = load_bT_bf16(xbf, DM_, row0 + nt * 16 + (lane & 15), kc, lane);
                acc[nt] = wmma_bf16(af, bf, acc[nt]);
            }
        }
        // lane holds fixed s, 8 consecutive d -> one 16B store per tile
        int d0 = wave * 16 + rsel;              // head-local dim base
        float bb[8];
#pragma unroll
        for (int i = 0; i < 8; ++i) bb[i] = bias[h * 64 + d0 + i];
#pragma unroll
        for (int nt = 0; nt < 4; ++nt) {
            int s = (row0 + nt * 16 + (lane & 15)) & (S_ - 1);
            pack_u p;
#pragma unroll
            for (int j = 0; j < 4; ++j)
                p.d[j] = pack2bf(acc[nt][2 * j] + bb[2 * j], acc[nt][2 * j + 1] + bb[2 * j + 1]);
            *(uint4*)(out + ((((size_t)b * H_ + h) * S_ + s) * D_) + d0) = p.q;
        }
    } else {
        // ---- V: A = x (M = seq), B = w (N = dim) -> store transposed [d][s]
        int arow = row0 + wave * 16 + (lane & 15);          // sequence row
        for (int kc = 0; kc < DM_; kc += 32) {
            v16bf af = load_a_bf16(xbf, DM_, arow, kc, lane);
#pragma unroll
            for (int nt = 0; nt < 4; ++nt) {
                v16bf bf = load_bT_bf16(w, DM_, h * 64 + nt * 16 + (lane & 15), kc, lane);
                acc[nt] = wmma_bf16(af, bf, acc[nt]);
            }
        }
        // lane holds fixed d, 8 consecutive s -> one 16B store per tile
        int s0 = (row0 + wave * 16 + rsel) & (S_ - 1);
#pragma unroll
        for (int nt = 0; nt < 4; ++nt) {
            int d = nt * 16 + (lane & 15);
            float bb = bias[h * 64 + d];
            pack_u p;
#pragma unroll
            for (int j = 0; j < 4; ++j)
                p.d[j] = pack2bf(acc[nt][2 * j] + bb, acc[nt][2 * j + 1] + bb);
            *(uint4*)(vt + ((((size_t)b * H_ + h) * D_ + d) * S_) + s0) = p.q;
        }
    }
}

// ------------- Stage 2: BigBird attention, flash online softmax -------------
// grid = B*H*NB blocks, 128 threads (4 waves). wave w owns rows [w*16, w*16+16).
__global__ __launch_bounds__(128) void bigbird_attn_kernel(
    const unsigned short* __restrict__ qg,
    const unsigned short* __restrict__ kg,
    const unsigned short* __restrict__ vtg,
    const int* __restrict__ rand_attn,
    float* __restrict__ out) {
    __shared__ float sS[64][65];
    __shared__ unsigned short sP[64][64];
    __shared__ float sM[64], sL[64], sA[64];

    int bid = blockIdx.x;
    int qb = bid & (NB_ - 1);
    int h  = (bid >> 6) & (H_ - 1);
    int b  = bid >> 9;
    const unsigned short* qh = qg  + (((size_t)b * H_ + h) * S_) * D_;
    const unsigned short* kh = kg  + (((size_t)b * H_ + h) * S_) * D_;
    const unsigned short* vh = vtg + (((size_t)b * H_ + h) * D_) * S_;   // [d][s]

    int tid = threadIdx.x, wv = tid >> 5, lane = tid & 31;
    int rsel = (lane < 16) ? 0 : 8;

    if (tid < 64) { sM[tid] = -1e30f; sL[tid] = 0.0f; }

    // preload Q fragments (rows fixed for this block)
    int qrow = qb * 64 + wv * 16 + (lane & 15);
    v16bf qf0 = load_a_bf16(qh, D_, qrow, 0, lane);
    v16bf qf1 = load_a_bf16(qh, D_, qrow, 32, lane);

    v8f o[4] = {v8f{}, v8f{}, v8f{}, v8f{}};

    // key-block list
    bool full = (qb == 0) || (qb == NB_ - 1);
    int list[8]; int nkb;
    if (full) { nkb = NB_; }
    else {
        int rrow = (qb == 1) ? 0 : ((qb == NB_ - 2) ? (NB_ - 3) : (qb - 1));
        int r0 = rand_attn[(h * (NB_ - 2) + rrow) * R_ + 0];
        int r1 = rand_attn[(h * (NB_ - 2) + rrow) * R_ + 1];
        int r2 = rand_attn[(h * (NB_ - 2) + rrow) * R_ + 2];
        if (qb == 1) {
            list[0] = 0; list[1] = 1; list[2] = 2; list[3] = NB_ - 1;
            list[4] = r0; list[5] = r1; list[6] = r2; nkb = 7;
        } else if (qb == NB_ - 2) {
            list[0] = 0; list[1] = NB_ - 3; list[2] = NB_ - 2; list[3] = NB_ - 1;
            list[4] = r0; list[5] = r1; list[6] = r2; nkb = 7;
        } else {
            list[0] = 0; list[1] = qb - 1; list[2] = qb; list[3] = qb + 1;
            list[4] = r0; list[5] = r1; list[6] = r2; list[7] = NB_ - 1; nkb = 8;
        }
    }
    __syncthreads();

    for (int it = 0; it < nkb; ++it) {
        int kb = full ? it : list[it];
        int kbase = kb * 64;

        // ---- scores S = (Q K^T) * scale, written to LDS ----
#pragma unroll
        for (int nt = 0; nt < 4; ++nt) {
            v8f s = v8f{};
            int keyrow = kbase + nt * 16 + (lane & 15);
            s = wmma_bf16(qf0, load_bT_bf16(kh, D_, keyrow, 0, lane), s);
            s = wmma_bf16(qf1, load_bT_bf16(kh, D_, keyrow, 32, lane), s);
#pragma unroll
            for (int i = 0; i < 8; ++i)
                sS[wv * 16 + i + rsel][nt * 16 + (lane & 15)] = s[i] * 0.125f;
        }
        __syncthreads();

        // ---- online softmax update, P tile to LDS (bf16) ----
        if (tid < 64) {
            float mold = sM[tid], mnew = mold;
#pragma unroll 8
            for (int c = 0; c < 64; ++c) mnew = fmaxf(mnew, sS[tid][c]);
            float alpha = __expf(mold - mnew);
            float lsum = 0.0f;
#pragma unroll 4
            for (int c = 0; c < 64; c += 2) {
                float p0 = __expf(sS[tid][c] - mnew);
                float p1 = __expf(sS[tid][c + 1] - mnew);
                lsum += p0 + p1;
                *(uint32_t*)&sP[tid][c] = pack2bf(p0, p1);
            }
            sM[tid] = mnew;
            sL[tid] = sL[tid] * alpha + lsum;
            sA[tid] = alpha;
        }
        __syncthreads();

        // ---- rescale O, accumulate P @ V ----
        float al[8];
#pragma unroll
        for (int i = 0; i < 8; ++i) al[i] = sA[wv * 16 + i + rsel];
#pragma unroll
        for (int nt = 0; nt < 4; ++nt)
#pragma unroll
            for (int i = 0; i < 8; ++i) o[nt][i] *= al[i];

        int prow = wv * 16 + (lane & 15);
        v16bf pf0 = load_a_bf16(&sP[0][0], 64, prow, 0, lane);
        v16bf pf1 = load_a_bf16(&sP[0][0], 64, prow, 32, lane);
#pragma unroll
        for (int nt = 0; nt < 4; ++nt) {
            int d = nt * 16 + (lane & 15);
            o[nt] = wmma_bf16(pf0, load_bT_bf16(vh, S_, d, kbase, lane), o[nt]);
            o[nt] = wmma_bf16(pf1, load_bT_bf16(vh, S_, d, kbase + 32, lane), o[nt]);
        }
        __syncthreads();
    }

    // ---- normalize and store: out[b][s][h*64+d] ----
    float linv[8];
#pragma unroll
    for (int i = 0; i < 8; ++i) linv[i] = 1.0f / sL[wv * 16 + i + rsel];
#pragma unroll
    for (int nt = 0; nt < 4; ++nt) {
        int d = nt * 16 + (lane & 15);
#pragma unroll
        for (int i = 0; i < 8; ++i) {
            int s = qb * 64 + wv * 16 + i + rsel;
            out[(((size_t)b * S_ + s) * (H_ * D_)) + h * 64 + d] = o[nt][i] * linv[i];
        }
    }
}

extern "C" void kernel_launch(void* const* d_in, const int* in_sizes, int n_in,
                              void* d_out, int out_size, void* d_ws, size_t ws_size,
                              hipStream_t stream) {
    const float* x  = (const float*)d_in[0];
    const float* wq = (const float*)d_in[1];
    const float* bq = (const float*)d_in[2];
    const float* wk = (const float*)d_in[3];
    const float* bk = (const float*)d_in[4];
    const float* wv = (const float*)d_in[5];
    const float* bv = (const float*)d_in[6];
    const int*   ra = (const int*)d_in[7];
    float* out = (float*)d_out;

    const size_t per = (size_t)B_ * H_ * S_ * D_;   // 4,194,304 bf16 elements
    unsigned short* qws  = (unsigned short*)d_ws;
    unsigned short* kws  = qws + per;
    unsigned short* vtws = kws + per;
    unsigned short* xbf  = vtws + per;               // B*S*DM = per
    unsigned short* wtws = xbf + per;                // 3*512*512

    conv_x_kernel<<<(B_ * S_ * DM_) / (256 * 8), 256, 0, stream>>>(x, xbf);
    conv_w_kernel<<<(3 * DM_ * DM_ / 2) / 256, 256, 0, stream>>>(wq, wk, wv, wtws);

    dim3 g1((B_ * S_) / 64, H_, 3);
    qkv_proj_kernel<<<g1, 128, 0, stream>>>(xbf, wtws, bq, bk, bv, qws, kws, vtws);

    dim3 g2(B_ * H_ * NB_);
    bigbird_attn_kernel<<<g2, 128, 0, stream>>>(qws, kws, vtws, ra, out);
}
